// SelfAttentionRope_67284957659772
// MI455X (gfx1250) — matrix operand order repaired and checked by reading, not compiled, not run on hardware
//
#include <hip/hip_runtime.h>
#include <hip/hip_fp16.h>

typedef __attribute__((ext_vector_type(16))) _Float16 v16h;
typedef __attribute__((ext_vector_type(8)))  _Float16 h8;
typedef __attribute__((ext_vector_type(8)))  float    v8f;

#define HEADS   16
#define DH      64
#define T_SEQ   2048
#define D_MODEL 1024
#define BATCH   4

#ifndef USE_ASYNC_LDS
#define USE_ASYNC_LDS 1
#endif
#ifndef USE_DPP_RED
#define USE_DPP_RED 1
#endif

// ---------------------------------------------------------------------------
// Async global->LDS DMA (gfx1250, ASYNCcnt-tracked)
// ---------------------------------------------------------------------------
__device__ __forceinline__ void async_copy_b128(unsigned int lds_off, const void* gaddr) {
#if USE_ASYNC_LDS
  asm volatile("global_load_async_to_lds_b128 %0, %1, off"
               :: "v"(lds_off), "v"((unsigned long long)(uintptr_t)gaddr)
               : "memory");
#else
  *(h8*)(uintptr_t)lds_off = *(const h8*)gaddr;  // fallback: sync copy
#endif
}
__device__ __forceinline__ void async_wait0() {
#if USE_ASYNC_LDS
  asm volatile("s_wait_asynccnt 0x0" ::: "memory");
#endif
}
__device__ __forceinline__ unsigned int lds_off32(const void* p) {
  return (unsigned int)(uintptr_t)p;
}

// ---------------------------------------------------------------------------
// 16-lane max reduction via DPP (no LDS, no DScnt stalls)
// ---------------------------------------------------------------------------
__device__ __forceinline__ float red_max16(float x) {
#if USE_DPP_RED
  int t;
  t = __builtin_amdgcn_update_dpp(0, __float_as_int(x), 0xB1, 0xF, 0xF, false);  // quad xor1
  x = fmaxf(x, __int_as_float(t));
  t = __builtin_amdgcn_update_dpp(0, __float_as_int(x), 0x4E, 0xF, 0xF, false);  // quad xor2
  x = fmaxf(x, __int_as_float(t));
  t = __builtin_amdgcn_update_dpp(0, __float_as_int(x), 0x141, 0xF, 0xF, false); // row_half_mirror
  x = fmaxf(x, __int_as_float(t));
  t = __builtin_amdgcn_update_dpp(0, __float_as_int(x), 0x140, 0xF, 0xF, false); // row_mirror
  x = fmaxf(x, __int_as_float(t));
  return x;
#else
  for (int off = 1; off < 16; off <<= 1) x = fmaxf(x, __shfl_xor(x, off, 16));
  return x;
#endif
}

// ---------------------------------------------------------------------------
// WMMA fragment loaders from LDS (CDNA5 wave32 layouts)
// ---------------------------------------------------------------------------
__device__ __forceinline__ v16h load_frag_a(const _Float16* rowk, int lh) {
  h8 lo = *(const h8*)(rowk + lh * 8);
  h8 hi = *(const h8*)(rowk + lh * 8 + 16);
  v16h a;
#pragma unroll
  for (int e = 0; e < 8; ++e) { a[e] = lo[e]; a[8 + e] = hi[e]; }
  return a;
}
__device__ __forceinline__ v16h load_frag_b(const _Float16* rowk, int lh) {
  h8 lo = *(const h8*)(rowk + lh * 8);
  h8 hi = *(const h8*)(rowk + lh * 8 + 16);
  v16h b;
#pragma unroll
  for (int e = 0; e < 8; ++e) { b[e] = lo[e]; b[8 + e] = hi[e]; }
  return b;
}

// ---------------------------------------------------------------------------
// fp32 -> f16, 4 elems/thread
// ---------------------------------------------------------------------------
__global__ void __launch_bounds__(256) cvt_f32_f16(const float* __restrict__ in,
                                                   _Float16* __restrict__ out, int n4) {
  int i = blockIdx.x * blockDim.x + threadIdx.x;
  if (i < n4) {
    float4 v = ((const float4*)in)[i];
    union { _Float16 h[4]; uint2 u; } p;
    p.h[0] = (_Float16)v.x; p.h[1] = (_Float16)v.y;
    p.h[2] = (_Float16)v.z; p.h[3] = (_Float16)v.w;
    ((uint2*)out)[i] = p.u;
  }
}

// ---------------------------------------------------------------------------
// fp32 RxC -> f16 CxR (tiled transpose); done once so GEMM B-tiles stage
// row-major via async DMA with no in-kernel transpose.
// ---------------------------------------------------------------------------
__global__ void __launch_bounds__(256) cvt_transpose(const float* __restrict__ in,
                                                     _Float16* __restrict__ out,
                                                     int R, int C) {
  __shared__ _Float16 tt[64][65];
  int r0 = blockIdx.y * 64, c0 = blockIdx.x * 64;
  int cl = threadIdx.x & 63, q = threadIdx.x >> 6;
#pragma unroll
  for (int i = 0; i < 16; ++i)
    tt[q + i * 4][cl] = (_Float16)in[(size_t)(r0 + q + i * 4) * C + c0 + cl];
  __syncthreads();
#pragma unroll
  for (int i = 0; i < 16; ++i)
    out[(size_t)(c0 + q + i * 4) * R + r0 + cl] = tt[cl][q + i * 4];
}

// ---------------------------------------------------------------------------
// WMMA GEMM: C = A[M,K] * BT[N,K]^T + bias. 64x64 tile, K-chunks of 32,
// double-buffered async staging, one barrier per iteration.
// ROPE=true: QKV projection epilogue: RoPE in-register, q pre-scaled by
// 1/sqrt(dh), V written transposed per head (vT[bh][d][t]).
// ---------------------------------------------------------------------------
template <bool ROPE>
__global__ void __launch_bounds__(128) gemm_wmma(
    const _Float16* __restrict__ A, const _Float16* __restrict__ BT,
    const float* __restrict__ bias, float* __restrict__ Cf,
    _Float16* __restrict__ Qo, _Float16* __restrict__ Ko, _Float16* __restrict__ Vo,
    int M, int N, int K) {
  __shared__ _Float16 sA[2][64][48];
  __shared__ _Float16 sBt[2][64][48];

  const int tid  = threadIdx.x;
  const int wave = tid >> 5, lane = tid & 31;
  const int ln = lane & 15, lh = lane >> 4;
  const int n0 = blockIdx.x * 64, m0 = blockIdx.y * 64;
  const int row = tid >> 1, seg = (tid & 1) * 16;

  auto stage = [&](int buf, int k0) {
    const _Float16* ga = A + (size_t)(m0 + row) * K + k0 + seg;
    unsigned int la = lds_off32(&sA[buf][row][seg]);
    async_copy_b128(la, ga);
    async_copy_b128(la + 16, ga + 8);
    const _Float16* gb = BT + (size_t)(n0 + row) * K + k0 + seg;
    unsigned int lb = lds_off32(&sBt[buf][row][seg]);
    async_copy_b128(lb, gb);
    async_copy_b128(lb + 16, gb + 8);
  };

  v8f acc[4];
  v8f zero = {};
#pragma unroll
  for (int j = 0; j < 4; ++j) acc[j] = zero;

  stage(0, 0);
  for (int k0 = 0; k0 < K; k0 += 32) {
    int buf = (k0 >> 5) & 1;
    async_wait0();
    __syncthreads();
    if (k0 + 32 < K) stage(buf ^ 1, k0 + 32);
    v16h a = load_frag_a(&sA[buf][wave * 16 + ln][0], lh);
#pragma unroll
    for (int j = 0; j < 4; ++j) {
      v16h b = load_frag_b(&sBt[buf][j * 16 + ln][0], lh);
      acc[j] = __builtin_amdgcn_wmma_f32_16x16x32_f16(false, a, false, b,
                                                      (short)0, acc[j], false, false);
    }
  }

  float vv[4][8];
#pragma unroll
  for (int j = 0; j < 4; ++j) {
    float bj = bias[n0 + j * 16 + ln];
#pragma unroll
    for (int r = 0; r < 8; ++r) vv[j][r] = acc[j][r] + bj;
  }

  if (!ROPE) {
#pragma unroll
    for (int r = 0; r < 8; ++r) {
      int gr = m0 + wave * 16 + r + lh * 8;
#pragma unroll
      for (int j = 0; j < 4; ++j)
        Cf[(size_t)gr * N + n0 + j * 16 + ln] = vv[j][r];
    }
  } else {
    const int third = n0 / D_MODEL;           // 0=q, 1=k, 2=v
    const int hh    = (n0 % D_MODEL) >> 6;    // head (tile == one head)
    if (third < 2) {
      _Float16* dst = (third == 0) ? Qo : Ko;
      const float oscale = (third == 0) ? 0.125f : 1.0f;  // fold 1/sqrt(64) into q
#pragma unroll
      for (int r = 0; r < 8; ++r) {
        int gr = m0 + wave * 16 + r + lh * 8;
        int b  = gr >> 11;
        int t  = gr & (T_SEQ - 1);
        float res[4];
#pragma unroll
        for (int jj = 0; jj < 2; ++jj) {
          int d0 = jj * 16 + ln;  // rotation-pair index 0..31
          float inv = __expf(-(float)d0 * (9.210340371976184f / 32.0f));
          float ang = (float)t * inv;
          float sn, cs;
          __sincosf(ang, &sn, &cs);
          float x1 = vv[jj][r], x2 = vv[jj + 2][r];
          res[jj]     = (x1 * cs - x2 * sn) * oscale;
          res[jj + 2] = (x1 * sn + x2 * cs) * oscale;
        }
        _Float16* o = dst + ((size_t)(b * HEADS + hh) * T_SEQ + t) * DH;
#pragma unroll
        for (int j = 0; j < 4; ++j) o[j * 16 + ln] = (_Float16)res[j];
      }
    } else {
      // V: store transposed per head, vectorized 8-wide along t
      int gr0 = m0 + wave * 16 + lh * 8;
      int b   = gr0 >> 11;
      int t0  = gr0 & (T_SEQ - 1);
#pragma unroll
      for (int j = 0; j < 4; ++j) {
        int d = j * 16 + ln;
        h8 pk;
#pragma unroll
        for (int r = 0; r < 8; ++r) pk[r] = (_Float16)vv[j][r];
        *(h8*)(Vo + ((size_t)(b * HEADS + hh) * DH + d) * T_SEQ + t0) = pk;
      }
    }
  }
}

// ---------------------------------------------------------------------------
// Flash attention: block = (b, h, 64 q-rows); 4 waves. All tiles staged by
// async LDS DMA (Q once; K and V^T double-buffered, prefetch overlaps
// compute). Row-sum of P computed by WMMA against an all-ones B fragment.
// ---------------------------------------------------------------------------
__global__ void __launch_bounds__(128) attn_wmma(const _Float16* __restrict__ Q,
                                                 const _Float16* __restrict__ K,
                                                 const _Float16* __restrict__ Vt,
                                                 _Float16* __restrict__ Ctx) {
  __shared__ _Float16 sQ[64][72];
  __shared__ _Float16 sK[2][64][72];   // row-major == Bt layout for Q*K^T
  __shared__ _Float16 sVt[2][64][72];  // sVt[d][key] (pre-transposed in memory)
  __shared__ _Float16 sP[64][72];

  const int tid  = threadIdx.x;
  const int wave = tid >> 5, lane = tid & 31;
  const int ln = lane & 15, lh = lane >> 4;
  const int bh = blockIdx.y;
  const int b = bh >> 4, h = bh & 15;
  const int q0 = blockIdx.x * 64;
  const int row = tid >> 1, seg = (tid & 1) * 32;

  const _Float16* Qp  = Q  + ((size_t)bh * T_SEQ + q0) * DH;
  const _Float16* Kp  = K  + (size_t)bh * T_SEQ * DH;
  const _Float16* Vtp = Vt + (size_t)bh * DH * T_SEQ;

  {  // stage Q once
    const _Float16* g = Qp + (size_t)row * DH + seg;
    unsigned int l = lds_off32(&sQ[row][seg]);
#pragma unroll
    for (int i = 0; i < 4; ++i) async_copy_b128(l + i * 16, g + i * 8);
  }
  auto stage_kv = [&](int buf, int kb) {
    const _Float16* gk = Kp + (size_t)(kb * 64 + row) * DH + seg;
    unsigned int lk = lds_off32(&sK[buf][row][seg]);
#pragma unroll
    for (int i = 0; i < 4; ++i) async_copy_b128(lk + i * 16, gk + i * 8);
    const _Float16* gv = Vtp + (size_t)row * T_SEQ + kb * 64 + seg;
    unsigned int lv = lds_off32(&sVt[buf][row][seg]);
#pragma unroll
    for (int i = 0; i < 4; ++i) async_copy_b128(lv + i * 16, gv + i * 8);
  };
  stage_kv(0, 0);

  v16h vone;
#pragma unroll
  for (int e = 0; e < 16; ++e) vone[e] = (_Float16)1.0f;

  float mrow[8], alpha[8];
  v8f oacc[4], oacc_l;
  v8f zero = {};
#pragma unroll
  for (int j = 0; j < 4; ++j) oacc[j] = zero;
  oacc_l = zero;
#pragma unroll
  for (int r = 0; r < 8; ++r) mrow[r] = -1.0e30f;

  const int NKB = T_SEQ / 64;
  for (int kb = 0; kb < NKB; ++kb) {
    int buf = kb & 1;
    async_wait0();
    __syncthreads();
    if (kb + 1 < NKB) stage_kv(buf ^ 1, kb + 1);

    // ---- S = Q * K^T (pre-scaled via Q) ----
    v8f s[4];
#pragma unroll
    for (int j = 0; j < 4; ++j) s[j] = zero;
#pragma unroll
    for (int ks = 0; ks < 2; ++ks) {
      v16h a = load_frag_a(&sQ[wave * 16 + ln][ks * 32], lh);
#pragma unroll
      for (int j = 0; j < 4; ++j) {
        v16h bfr = load_frag_b(&sK[buf][j * 16 + ln][ks * 32], lh);
        s[j] = __builtin_amdgcn_wmma_f32_16x16x32_f16(false, a, false, bfr,
                                                      (short)0, s[j], false, false);
      }
    }

    // ---- online softmax: DPP max, exp, alpha; row-sum handled by WMMA ----
#pragma unroll
    for (int r = 0; r < 8; ++r) {
      float mx = red_max16(fmaxf(fmaxf(s[0][r], s[1][r]), fmaxf(s[2][r], s[3][r])));
      float nm = fmaxf(mrow[r], mx);
      alpha[r] = __expf(mrow[r] - nm);
      mrow[r]  = nm;
#pragma unroll
      for (int j = 0; j < 4; ++j) s[j][r] = __expf(s[j][r] - nm);
    }
#pragma unroll
    for (int jd = 0; jd < 4; ++jd)
#pragma unroll
      for (int r = 0; r < 8; ++r) oacc[jd][r] *= alpha[r];
#pragma unroll
    for (int r = 0; r < 8; ++r) oacc_l[r] *= alpha[r];

    // ---- P to LDS (C-layout -> A-layout; same-wave rows only) ----
#pragma unroll
    for (int j = 0; j < 4; ++j)
#pragma unroll
      for (int r = 0; r < 8; ++r)
        sP[wave * 16 + r + lh * 8][j * 16 + ln] = (_Float16)s[j][r];

    // ---- O += P*V ; l += P*ones (same A fragments) ----
#pragma unroll
    for (int ks = 0; ks < 2; ++ks) {
      v16h a = load_frag_a(&sP[wave * 16 + ln][ks * 32], lh);
      oacc_l = __builtin_amdgcn_wmma_f32_16x16x32_f16(false, a, false, vone,
                                                      (short)0, oacc_l, false, false);
#pragma unroll
      for (int jd = 0; jd < 4; ++jd) {
        v16h bfr = load_frag_b(&sVt[buf][jd * 16 + ln][ks * 32], lh);
        oacc[jd] = __builtin_amdgcn_wmma_f32_16x16x32_f16(false, a, false, bfr,
                                                          (short)0, oacc[jd], false, false);
      }
    }
  }

  // ---- normalize and write context (B,T,D) as f16 ----
#pragma unroll
  for (int r = 0; r < 8; ++r) {
    int m = wave * 16 + r + lh * 8;
    int t = q0 + m;
    float rl = 1.0f / oacc_l[r];
#pragma unroll
    for (int jd = 0; jd < 4; ++jd) {
      float val = oacc[jd][r] * rl;
      Ctx[((size_t)b * T_SEQ + t) * D_MODEL + h * DH + jd * 16 + ln] = (_Float16)val;
    }
  }
}

// ---------------------------------------------------------------------------
extern "C" void kernel_launch(void* const* d_in, const int* in_sizes, int n_in,
                              void* d_out, int out_size, void* d_ws, size_t ws_size,
                              hipStream_t stream) {
  const float* x     = (const float*)d_in[0];
  const float* Wqkv  = (const float*)d_in[1];
  const float* bqkv  = (const float*)d_in[2];
  const float* Wproj = (const float*)d_in[3];
  const float* bproj = (const float*)d_in[4];
  float* out = (float*)d_out;

  const size_t BT = (size_t)BATCH * T_SEQ;  // 8192
  _Float16* p = (_Float16*)d_ws;
  size_t o = 0;
  _Float16* xh   = p + o; o += BT * D_MODEL;
  _Float16* WqT  = p + o; o += (size_t)D_MODEL * 3 * D_MODEL;  // Wqkv^T (3072x1024)
  _Float16* WpT  = p + o; o += (size_t)D_MODEL * D_MODEL;      // Wproj^T
  _Float16* qh   = p + o; o += BT * D_MODEL;   // q roped+scaled, [bh][t][d]
  _Float16* kh   = p + o; o += BT * D_MODEL;   // k roped,        [bh][t][d]
  _Float16* vT   = p + o; o += BT * D_MODEL;   // v transposed,   [bh][d][t]
  _Float16* ctx  = p + o; o += BT * D_MODEL;   // attention output (B,T,D)

  int n1 = (int)(BT * D_MODEL) / 4;
  cvt_f32_f16<<<(n1 + 255) / 256, 256, 0, stream>>>(x, xh, n1);
  cvt_transpose<<<dim3(3 * D_MODEL / 64, D_MODEL / 64), 256, 0, stream>>>(
      Wqkv, WqT, D_MODEL, 3 * D_MODEL);
  cvt_transpose<<<dim3(D_MODEL / 64, D_MODEL / 64), 256, 0, stream>>>(
      Wproj, WpT, D_MODEL, D_MODEL);

  gemm_wmma<true><<<dim3(3 * D_MODEL / 64, BT / 64), 128, 0, stream>>>(
      xh, WqT, bqkv, nullptr, qh, kh, vT, (int)BT, 3 * D_MODEL, D_MODEL);

  attn_wmma<<<dim3(T_SEQ / 64, BATCH * HEADS), 128, 0, stream>>>(qh, kh, vT, ctx);

  gemm_wmma<false><<<dim3(D_MODEL / 64, BT / 64), 128, 0, stream>>>(
      ctx, WpT, bproj, out, nullptr, nullptr, nullptr, (int)BT, D_MODEL, D_MODEL);
}
